// SingleHeadCausalSelfAttention_61830349193294
// MI455X (gfx1250) — matrix operand order repaired and checked by reading, not compiled, not run on hardware
//
#include <hip/hip_runtime.h>

#define B_  8
#define T_  2048
#define C_  768
#define M_  (B_ * T_)          // 16384 rows total
#define QK_SCALE 0.0360843918243516f   // 1/sqrt(768)

typedef unsigned short u16;
typedef unsigned int   u32;

typedef __attribute__((ext_vector_type(16))) __bf16 v16bf;
typedef __attribute__((ext_vector_type(8)))  __bf16 v8bf;
typedef __attribute__((ext_vector_type(8)))  float  v8f;

// ---------- helpers ----------
__device__ __forceinline__ u16 f2bf(float f) {
    u32 u = __builtin_bit_cast(u32, f);
    u += 0x7FFFu + ((u >> 16) & 1u);        // round-to-nearest-even
    return (u16)(u >> 16);
}
__device__ __forceinline__ float bf2f(u16 h) {
    return __builtin_bit_cast(float, (u32)h << 16);
}
__device__ __forceinline__ v8f wmma_bf16(v16bf a, v16bf b, v8f c) {
    return __builtin_amdgcn_wmma_f32_16x16x32_bf16(
        /*neg_a=*/false, a, /*neg_b=*/false, b,
        /*c_mod=*/(short)0, c, /*reuse_a=*/false, /*reuse_b=*/false);
}
__device__ __forceinline__ v16bf join8(v8bf lo, v8bf hi) {
    union { v16bf v; v8bf h[2]; } u;
    u.h[0] = lo; u.h[1] = hi;
    return u.v;
}
__device__ __forceinline__ v8f vzero8() {
    v8f z = {0.f, 0.f, 0.f, 0.f, 0.f, 0.f, 0.f, 0.f};
    return z;
}

// ---------- 1) fp32 -> bf16 convert ----------
__global__ void cvt_f32_bf16(const float* __restrict__ src, u16* __restrict__ dst, int n) {
    int i = blockIdx.x * blockDim.x + threadIdx.x;
    if (i < n) dst[i] = f2bf(src[i]);
}

// ---------- 2) QKV projection GEMM (double-buffered LDS B tile) ----------
// out[m][n] = sum_k xb[m][k] * W[n][k].  Wave owns a distinct 16-row m-tile;
// all 8 waves share the 64-wide n-slice, whose 64x32 B-tile is staged in LDS.
// Double buffering: next tile is prefetched to registers before the WMMAs
// (latency hidden behind matrix ops), then stored to the alternate buffer;
// one barrier per k-step.
// mat 0 -> Qb (scaled, [t][d]) ; mat 1 -> Kb ([s][d]) ; mat 2 -> Vt ([b][d][t])
__global__ void qkv_gemm_kernel(const u16* __restrict__ xb, const u16* __restrict__ wb,
                                u16* __restrict__ Qb, u16* __restrict__ Kb,
                                u16* __restrict__ Vt) {
    __shared__ __align__(32) u16 Bs[2][4 * 32 * 16];   // 2 x 4KB double buffer

    const int tid  = threadIdx.x;
    const int lane = tid & 31;
    const int wave = tid >> 5;
    const int hl   = lane >> 4;
    const int r    = lane & 15;

    const int m0  = (blockIdx.x * 8 + wave) * 16;
    const int n0  = blockIdx.y * 64;
    const int mat = blockIdx.z;

    const u16* W = wb + (size_t)mat * C_ * C_;

    // fill mapping: thread -> (n-tile j, fragment lane l, 8-elem half)
    const int fj   = tid >> 6;
    const int fl   = (tid >> 1) & 31;
    const int fp   = tid & 1;
    const int foff = (fj * 32 + fl) * 16 + fp * 8;
    const u16* wsrc = W + (size_t)(n0 + fj * 16 + (fl & 15)) * C_ + (fl >> 4) * 8 + fp * 16;

    // prologue: stage k=0 tile into buffer 0
    *(v8bf*)(&Bs[0][0] + foff) = *(const v8bf*)(wsrc);
    __syncthreads();

    v8f acc[4] = {vzero8(), vzero8(), vzero8(), vzero8()};
    const u16* arow = xb + (size_t)(m0 + r) * C_ + hl * 8;

    int cur = 0;
    for (int k = 0; k < C_; k += 32) {
        const bool more = (k + 32) < C_;
        const int  kn   = more ? (k + 32) : k;            // clamped: always in-bounds
        v8bf nxt = *(const v8bf*)(wsrc + kn);             // prefetch next B chunk

        v16bf a = join8(*(const v8bf*)(arow + k), *(const v8bf*)(arow + k + 16));
        const u16* base = &Bs[cur][0];
        acc[0] = wmma_bf16(a, *(const v16bf*)(base + (0 * 32 + lane) * 16), acc[0]);
        acc[1] = wmma_bf16(a, *(const v16bf*)(base + (1 * 32 + lane) * 16), acc[1]);
        acc[2] = wmma_bf16(a, *(const v16bf*)(base + (2 * 32 + lane) * 16), acc[2]);
        acc[3] = wmma_bf16(a, *(const v16bf*)(base + (3 * 32 + lane) * 16), acc[3]);

        if (more) *(v8bf*)(&Bs[cur ^ 1][0] + foff) = nxt; // fill alternate buffer
        cur ^= 1;
        __syncthreads();
    }

    const float scale = (mat == 0) ? QK_SCALE : 1.0f;
    #pragma unroll
    for (int j = 0; j < 4; ++j) {
        const int col = n0 + j * 16 + r;
        #pragma unroll
        for (int e = 0; e < 8; ++e) {
            const int row = m0 + e + 8 * hl;
            const u16 v = f2bf(acc[j][e] * scale);
            if (mat == 0) {
                Qb[(size_t)row * C_ + col] = v;
            } else if (mat == 1) {
                Kb[(size_t)row * C_ + col] = v;
            } else {
                const int bb = row >> 11;          // row / T_
                const int tl = row & (T_ - 1);     // row % T_
                Vt[((size_t)bb * C_ + col) * T_ + tl] = v;
            }
        }
    }
}

// ---------- 3) causal scores: S = Q K^T (LDS-staged Q tile) ----------
// All 8 waves share the same 16-row Q tile -> copied once to LDS (24KB,
// contiguous); waves then stream only K fragments from global.
__global__ void scores_kernel(const u16* __restrict__ Qb, const u16* __restrict__ Kb,
                              u16* __restrict__ Sb) {
    __shared__ __align__(32) u16 Qs[16 * C_];       // 24KB

    const int tid  = threadIdx.x;
    const int lane = tid & 31;
    const int wave = tid >> 5;
    const int hl   = lane >> 4;
    const int r    = lane & 15;

    const int t0 = blockIdx.y * 16;
    const int b  = blockIdx.z;

    // block-uniform causal skip: whole s-group above diagonal
    if (blockIdx.x * 128 > t0 + 15) return;

    // cooperative copy of Q tile (rows t0..t0+15 are contiguous in memory)
    {
        const u16* qbase = Qb + ((size_t)b * T_ + t0) * C_;
        for (int i = tid; i < (16 * C_) / 8; i += 256)
            *(v8bf*)(Qs + i * 8) = *(const v8bf*)(qbase + i * 8);
    }
    __syncthreads();

    const int s0 = (blockIdx.x * 8 + wave) * 16;
    if (s0 > t0 + 15) return;   // wave-uniform skip (no barriers after this)

    const u16* arow = Qs + r * C_ + hl * 16;
    const u16* krow = Kb + ((size_t)b * T_ + s0 + r) * C_ + hl * 16;

    v8f acc = vzero8();
    #pragma unroll 4
    for (int d = 0; d < C_; d += 32) {
        v16bf a = join8(*(const v8bf*)(arow + d), *(const v8bf*)(arow + d + 8));
        acc = wmma_bf16(a, *(const v16bf*)(krow + d), acc);
    }

    #pragma unroll
    for (int e = 0; e < 8; ++e) {
        const int t = t0 + e + 8 * hl;
        Sb[((size_t)b * T_ + t) * T_ + s0 + r] = f2bf(acc[e]);
    }
}

// ---------- 4) row softmax with causal mask (in place, bf16) ----------
__global__ void softmax_kernel(u16* __restrict__ Sb) {
    __shared__ float red[256];
    const int row = blockIdx.x;           // 0 .. B*T-1
    const int t   = row & (T_ - 1);
    const int tid = threadIdx.x;
    u16* S = Sb + (size_t)row * T_;

    float m = -3.0e38f;
    for (int s = tid; s <= t; s += 256) m = fmaxf(m, bf2f(S[s]));
    red[tid] = m; __syncthreads();
    for (int off = 128; off > 0; off >>= 1) {
        if (tid < off) red[tid] = fmaxf(red[tid], red[tid + off]);
        __syncthreads();
    }
    m = red[0]; __syncthreads();

    float l = 0.f;
    for (int s = tid; s <= t; s += 256) l += __expf(bf2f(S[s]) - m);
    red[tid] = l; __syncthreads();
    for (int off = 128; off > 0; off >>= 1) {
        if (tid < off) red[tid] += red[tid + off];
        __syncthreads();
    }
    const float inv = 1.0f / red[0];

    for (int s = tid; s < T_; s += 256) {
        if (s <= t) S[s] = f2bf(__expf(bf2f(S[s]) - m) * inv);
        else        S[s] = 0;   // zero masked region (also clears poison)
    }
}

// ---------- 5) O = P V (double-buffered LDS V^T tile) ----------
// Waves own distinct 16-row t-tiles but share the 64-wide d-slice: the 64x32
// V^T B-tile is double-buffered in LDS per s-step (prefetch to registers over
// the WMMAs, one barrier per step). Loop runs to the block's max causal bound;
// waves past their own bound skip the MMA work (wave-uniform branch) but keep
// participating in fills and barriers.
__global__ void pv_kernel(const u16* __restrict__ Sb, const u16* __restrict__ Vt,
                          float* __restrict__ out) {
    __shared__ __align__(32) u16 Vs[2][4 * 32 * 16];   // 2 x 4KB double buffer

    const int tid  = threadIdx.x;
    const int lane = tid & 31;
    const int wave = tid >> 5;
    const int hl   = lane >> 4;
    const int r    = lane & 15;

    const int t0 = (blockIdx.x * 8 + wave) * 16;
    const int d0 = blockIdx.y * 64;
    const int b  = blockIdx.z;

    // fill mapping: thread -> (d-tile j, fragment lane l, 8-elem half)
    const int fj   = tid >> 6;
    const int fl   = (tid >> 1) & 31;
    const int fp   = tid & 1;
    const int foff = (fj * 32 + fl) * 16 + fp * 8;
    const u16* vsrc = Vt + ((size_t)b * C_ + d0 + fj * 16 + (fl & 15)) * T_ + (fl >> 4) * 8 + fp * 16;

    const u16* prow = Sb + ((size_t)b * T_ + t0 + r) * T_ + hl * 8;

    v8f acc[4] = {vzero8(), vzero8(), vzero8(), vzero8()};

    const int s_end  = (t0 + 16 + 31) & ~31;                     // this wave's causal bound
    const int t0max  = (blockIdx.x * 8 + 7) * 16;
    const int s_stop = (t0max + 16 + 31) & ~31;                  // block-uniform bound

    // prologue: stage s=0 tile into buffer 0
    *(v8bf*)(&Vs[0][0] + foff) = *(const v8bf*)(vsrc);
    __syncthreads();

    int cur = 0;
    for (int s = 0; s < s_stop; s += 32) {
        const bool more = (s + 32) < s_stop;
        const int  sn   = more ? (s + 32) : s;                   // clamped: in-bounds
        v8bf nxt = *(const v8bf*)(vsrc + sn);                    // prefetch next V^T chunk

        if (s < s_end) {                                         // wave-uniform
            v16bf a = join8(*(const v8bf*)(prow + s), *(const v8bf*)(prow + s + 16));
            const u16* base = &Vs[cur][0];
            acc[0] = wmma_bf16(a, *(const v16bf*)(base + (0 * 32 + lane) * 16), acc[0]);
            acc[1] = wmma_bf16(a, *(const v16bf*)(base + (1 * 32 + lane) * 16), acc[1]);
            acc[2] = wmma_bf16(a, *(const v16bf*)(base + (2 * 32 + lane) * 16), acc[2]);
            acc[3] = wmma_bf16(a, *(const v16bf*)(base + (3 * 32 + lane) * 16), acc[3]);
        }

        if (more) *(v8bf*)(&Vs[cur ^ 1][0] + foff) = nxt;        // fill alternate buffer
        cur ^= 1;
        __syncthreads();
    }

    #pragma unroll
    for (int j = 0; j < 4; ++j) {
        const int col = d0 + j * 16 + r;
        #pragma unroll
        for (int e = 0; e < 8; ++e) {
            const int t = t0 + e + 8 * hl;
            out[((size_t)b * T_ + t) * C_ + col] = acc[j][e];
        }
    }
}

// ---------- launch ----------
extern "C" void kernel_launch(void* const* d_in, const int* in_sizes, int n_in,
                              void* d_out, int out_size, void* d_ws, size_t ws_size,
                              hipStream_t stream) {
    const float* x  = (const float*)d_in[0];
    const float* Wq = (const float*)d_in[1];
    const float* Wk = (const float*)d_in[2];
    const float* Wv = (const float*)d_in[3];
    float* out = (float*)d_out;

    // workspace carve-up (bf16 elements)
    u16* xb = (u16*)d_ws;                         // [M, C]
    u16* wb = xb + (size_t)M_ * C_;               // [3, C, C]
    u16* Qb = wb + (size_t)3 * C_ * C_;           // [B, T, C]  (scaled)
    u16* Kb = Qb + (size_t)M_ * C_;               // [B, T, C]
    u16* Vt = Kb + (size_t)M_ * C_;               // [B, C, T]  (transposed)
    u16* Sb = Vt + (size_t)M_ * C_;               // [B, T, T]  logits -> probs

    // 1) converts
    {
        const int nx = M_ * C_;
        cvt_f32_bf16<<<(nx + 255) / 256, 256, 0, stream>>>(x, xb, nx);
        const int nw = C_ * C_;
        cvt_f32_bf16<<<(nw + 255) / 256, 256, 0, stream>>>(Wq, wb,          nw);
        cvt_f32_bf16<<<(nw + 255) / 256, 256, 0, stream>>>(Wk, wb + nw,     nw);
        cvt_f32_bf16<<<(nw + 255) / 256, 256, 0, stream>>>(Wv, wb + 2 * nw, nw);
    }
    // 2) QKV projections: grid (m-tiles/8, n64-tiles, matrix)
    qkv_gemm_kernel<<<dim3(M_ / 16 / 8, C_ / 64, 3), 256, 0, stream>>>(xb, wb, Qb, Kb, Vt);
    // 3) causal scores
    scores_kernel<<<dim3(T_ / 16 / 8, T_ / 16, B_), 256, 0, stream>>>(Qb, Kb, Sb);
    // 4) softmax (one block per row)
    softmax_kernel<<<dim3(B_ * T_), 256, 0, stream>>>(Sb);
    // 5) P @ V
    pv_kernel<<<dim3(T_ / 16 / 8, C_ / 64, B_), 256, 0, stream>>>(Sb, Vt, out);
}
